// SpeechFormerCTC_32676111188520
// MI455X (gfx1250) — compile-verified, matrix operand order checked by reading, not run on hardware
//
#include <hip/hip_runtime.h>

// ---------------------------------------------------------------------------
// SpeechFormer forward for MI455X (gfx1250), wave32 + WMMA bf16.
// ---------------------------------------------------------------------------
// Shapes: B=16, C=60, MEL=128, T=400, FD=SD=512, POOL=64, H=8, DH=64, L=6, UD=256
// ---------------------------------------------------------------------------

typedef __attribute__((ext_vector_type(16))) __bf16 bf16x16;
typedef __attribute__((ext_vector_type(8)))  float  floatx8;

union FragU { uint4 u[2]; bf16x16 v; };

#define DEV static __device__ __forceinline__

DEV unsigned short f2bf(float x) {
  unsigned int u = __float_as_uint(x);
  return (unsigned short)((u + 0x7FFFu + ((u >> 16) & 1u)) >> 16);
}
DEV unsigned int pack2bf(float lo, float hi) {
  return (unsigned int)f2bf(lo) | ((unsigned int)f2bf(hi) << 16);
}
DEV floatx8 zero8() { floatx8 z = {0.f,0.f,0.f,0.f,0.f,0.f,0.f,0.f}; return z; }

// A-fragment (and row-major-B-fragment) gather from LDS.
// 16-bit A 16x32 layout: lane<16 -> row, k=0..7 (dw0..3) & 16..23 (dw4..7);
// lane>=16 adds +8 to k. Two ds_load_b128 per fragment.
DEV bf16x16 lds_frag(const unsigned short* base, int strideHalfs, int rowBase, int kHalf) {
  int lane = threadIdx.x & 31;
  const unsigned short* p =
      base + (rowBase + (lane & 15)) * strideHalfs + kHalf + ((lane >> 4) << 3);
  FragU f;
  f.u[0] = *reinterpret_cast<const uint4*>(p);
  f.u[1] = *reinterpret_cast<const uint4*>(p + 16);
  return f.v;
}

// Pre-swizzled B-fragment from global memory: 8 contiguous dwords per lane.
DEV bf16x16 gmem_frag(const unsigned int* base, int nTile, int kStep, int nks) {
  int lane = threadIdx.x & 31;
  const unsigned int* p = base + ((((nTile * nks + kStep) << 5) + lane) << 3);
  FragU f;
  f.u[0] = *reinterpret_cast<const uint4*>(p);
  f.u[1] = *reinterpret_cast<const uint4*>(p + 4);
  return f.v;
}

DEV floatx8 wmma_bf16(bf16x16 a, bf16x16 b, floatx8 c) {
  return __builtin_amdgcn_wmma_f32_16x16x32_bf16(false, a, false, b, (short)0, c,
                                                 false, false);
}

DEV float redsum(float v, int maxMask) {
  for (int m = 1; m <= maxMask; m <<= 1) v += __shfl_xor(v, m, 32);
  return v;
}
DEV float redmax16(float v) {
  for (int m = 1; m <= 8; m <<= 1) v = fmaxf(v, __shfl_xor(v, m, 32));
  return v;
}

// ---------------------------------------------------------------------------
// Weight pre-swizzle: fp32 [K][N] -> bf16 B-fragment order (dwords).
// dst dword index g = (((nTile*nks + kStep)*32 + lane)*8 + dw)
// ---------------------------------------------------------------------------
__global__ __launch_bounds__(256) void prep_frag(const float* __restrict__ W,
                                                 unsigned int* __restrict__ dst,
                                                 int K, int N) {
  int g = blockIdx.x * 256 + threadIdx.x;
  int total = (K * N) >> 1;
  if (g >= total) return;
  int dw = g & 7, lane = (g >> 3) & 31;
  int t = g >> 8;
  int nks = K >> 5;
  int kStep = t % nks, nTile = t / nks;
  int n = nTile * 16 + (lane & 15);
  int kOff = (dw < 4 ? 2 * dw : 16 + 2 * (dw - 4)) + ((lane >> 4) << 3);
  int k = kStep * 32 + kOff;
  dst[g] = pack2bf(W[(size_t)k * N + n], W[(size_t)(k + 1) * N + n]);
}

__global__ void zero_kernel(float* p) { p[blockIdx.x * 256 + threadIdx.x] = 0.f; }

// ---------------------------------------------------------------------------
// Kernel 1: fused frame encoder.  grid = B*C = 960, 256 thr (8 waves).
// Per (b,c): for each 16-frame block: A=mel^T (bf16, LDS) x fe_w frags -> LN ->
// ReLU -> accumulate time-mean in registers.  Only seg[16,64,512] hits HBM.
// ---------------------------------------------------------------------------
__global__ __launch_bounds__(256) void encoder_kernel(
    const float* __restrict__ mel, const unsigned int* __restrict__ feF,
    const float* __restrict__ fe_b, const float* __restrict__ fe_g,
    const float* __restrict__ fe_beta, float* __restrict__ seg) {
  __shared__ unsigned short aT[16 * 128];   // transposed mel tile, bf16
  __shared__ float fr[16 * 512];            // frame tile pre-LN
  __shared__ float wacc[8][512];            // per-wave partial sums
  int bc = blockIdx.x;
  int tid = threadIdx.x, lane = tid & 31, w = tid >> 5, hf = lane >> 4;
  const float* melbc = mel + (size_t)bc * 128 * 400;
  float facc[16];
#pragma unroll
  for (int j = 0; j < 16; ++j) facc[j] = 0.f;

  for (int tb = 0; tb < 25; ++tb) {
    int t0 = tb * 16;
    // stage + transpose mel[m][t] -> aT[t][m] (bf16)
    for (int e = tid; e < 2048; e += 256) {
      int t = e & 15, m = e >> 4;
      const float* p = melbc + m * 400 + t0 + t;
      aT[t * 128 + m] = f2bf(*p);
      if (tb < 24) __builtin_prefetch(p + 16, 0, 1);  // global_prefetch next block
    }
    __syncthreads();
    // GEMM: each wave does 4 f-tiles, K=128 (4 WMMAs each)
    for (int i = 0; i < 4; ++i) {
      int ft = w + 8 * i;
      floatx8 c = zero8();
#pragma unroll
      for (int ks = 0; ks < 4; ++ks)
        c = wmma_bf16(lds_frag(aT, 128, 0, ks * 32), gmem_frag(feF, ft, ks, 4), c);
      int n = ft * 16 + (lane & 15);
      float bias = fe_b[n];
#pragma unroll
      for (int r = 0; r < 8; ++r) fr[(r + 8 * hf) * 512 + n] = c[r] + bias;
    }
    __syncthreads();
    // LN + ReLU + accumulate: wave w handles rows 2w, 2w+1
    for (int rr = 0; rr < 2; ++rr) {
      int row = 2 * w + rr;
      float vals[16], s = 0.f, ss = 0.f;
#pragma unroll
      for (int j = 0; j < 16; ++j) {
        float v = fr[row * 512 + lane + 32 * j];
        vals[j] = v; s += v; ss += v * v;
      }
      s = redsum(s, 16); ss = redsum(ss, 16);
      float mean = s * (1.f / 512.f);
      float inv = rsqrtf(ss * (1.f / 512.f) - mean * mean + 1e-5f);
#pragma unroll
      for (int j = 0; j < 16; ++j) {
        int f = lane + 32 * j;
        float v = (vals[j] - mean) * inv * fe_g[f] + fe_beta[f];
        facc[j] += fmaxf(v, 0.f);
      }
    }
    __syncthreads();
  }
#pragma unroll
  for (int j = 0; j < 16; ++j) wacc[w][lane + 32 * j] = facc[j];
  __syncthreads();
  int b = bc / 60, cc = bc % 60;
  float* segp = seg + ((size_t)b * 64 + cc) * 512;
  for (int f = tid; f < 512; f += 256) {
    float s = 0.f;
#pragma unroll
    for (int ww = 0; ww < 8; ++ww) s += wacc[ww][f];
    segp[f] = s * (1.f / 400.f);
  }
}

// ---------------------------------------------------------------------------
// Kernel 2: segment projection + LN.  grid = B*4 = 64 (16-row tiles).
// ---------------------------------------------------------------------------
__global__ __launch_bounds__(256) void segproj_kernel(
    const float* __restrict__ seg, const unsigned int* __restrict__ spF,
    const float* __restrict__ sp_b, const float* __restrict__ sp_g,
    const float* __restrict__ sp_beta, float* __restrict__ xbuf) {
  __shared__ unsigned short xo[16 * 512];
  __shared__ float y[16 * 512];
  int b = blockIdx.x >> 2, mt = blockIdx.x & 3;
  int tid = threadIdx.x, lane = tid & 31, w = tid >> 5, hf = lane >> 4;
  const float* src = seg + ((size_t)b * 64 + mt * 16) * 512;
  for (int e = tid; e < 16 * 512; e += 256) xo[e] = f2bf(src[e]);
  __syncthreads();
  for (int i = 0; i < 4; ++i) {
    int nt = w + 8 * i;
    floatx8 c = zero8();
#pragma unroll
    for (int ks = 0; ks < 16; ++ks)
      c = wmma_bf16(lds_frag(xo, 512, 0, ks * 32), gmem_frag(spF, nt, ks, 16), c);
    int n = nt * 16 + (lane & 15);
    float bias = sp_b[n];
#pragma unroll
    for (int r = 0; r < 8; ++r) y[(r + 8 * hf) * 512 + n] = c[r] + bias;
  }
  __syncthreads();
  float* xr = xbuf + ((size_t)b * 64 + mt * 16) * 512;
  for (int rr = 0; rr < 2; ++rr) {
    int row = 2 * w + rr;
    float vals[16], s = 0.f, ss = 0.f;
#pragma unroll
    for (int j = 0; j < 16; ++j) {
      float v = y[row * 512 + lane + 32 * j];
      vals[j] = v; s += v; ss += v * v;
    }
    s = redsum(s, 16); ss = redsum(ss, 16);
    float mean = s * (1.f / 512.f);
    float inv = rsqrtf(ss * (1.f / 512.f) - mean * mean + 1e-5f);
#pragma unroll
    for (int j = 0; j < 16; ++j) {
      int f = lane + 32 * j;
      xr[row * 512 + f] = (vals[j] - mean) * inv * sp_g[f] + sp_beta[f];
    }
  }
}

// ---------------------------------------------------------------------------
// Kernel 3a: QKV + attention for one (batch, head).  grid = B*H = 128.
// LDS: xin bf16[64][512] | q[64][64] | k[64][64] | v^T[64][64]  (88 KB dynamic)
// ---------------------------------------------------------------------------
__global__ __launch_bounds__(256) void attn_kernel(
    const float* __restrict__ xbuf, const unsigned int* __restrict__ qkvF,
    const float* __restrict__ qkv_b, unsigned short* __restrict__ obuf, int layer) {
  extern __shared__ char smem[];
  unsigned short* xin = (unsigned short*)smem;       // 64*512
  unsigned short* qL  = xin + 64 * 512;              // 64*64
  unsigned short* kL  = qL + 64 * 64;
  unsigned short* vtL = kL + 64 * 64;                // transposed V: [d][seq]
  int b = blockIdx.x >> 3, head = blockIdx.x & 7;
  int tid = threadIdx.x, lane = tid & 31, w = tid >> 5, hf = lane >> 4;
  const float* xb = xbuf + (size_t)b * 64 * 512;
  const unsigned int* qf = qkvF + (size_t)layer * (512 * 1536 / 2);
  const float* qb = qkv_b + layer * 1536;

  for (int e = tid; e < 64 * 512; e += 256) xin[e] = f2bf(xb[e]);
  __syncthreads();

  // QKV GEMM restricted to this head's 12 column tiles (4 q, 4 k, 4 v) x 4 row tiles
  for (int i = 0; i < 6; ++i) {
    int tile = w + 8 * i;               // 0..47
    int mt = tile / 12, ntL = tile % 12;
    int sec = ntL >> 2;                 // 0=q 1=k 2=v
    int ntG = sec * 32 + head * 4 + (ntL & 3);
    floatx8 c = zero8();
#pragma unroll
    for (int ks = 0; ks < 16; ++ks)
      c = wmma_bf16(lds_frag(xin, 512, mt * 16, ks * 32),
                    gmem_frag(qf, ntG, ks, 16), c);
    int n = ntG * 16 + (lane & 15);
    float bias = qb[n];
    int d = (ntL & 3) * 16 + (lane & 15);
#pragma unroll
    for (int r = 0; r < 8; ++r) {
      int seq = mt * 16 + r + 8 * hf;
      unsigned short v = f2bf(c[r] + bias);
      if (sec == 0)      qL[seq * 64 + d] = v;
      else if (sec == 1) kL[seq * 64 + d] = v;
      else               vtL[d * 64 + seq] = v;       // store V transposed
    }
  }
  __syncthreads();

  // Attention: waves 0..3 each own one 16-query tile (uniform branch: EXEC full)
  if (w < 4) {
    int mt = w;
    floatx8 s[4];
#pragma unroll
    for (int nt = 0; nt < 4; ++nt) {
      s[nt] = zero8();
#pragma unroll
      for (int ks = 0; ks < 2; ++ks)
        s[nt] = wmma_bf16(lds_frag(qL, 64, mt * 16, ks * 32),
                          lds_frag(kL, 64, nt * 16, ks * 32), s[nt]);
    }
    unsigned short* P = qL;             // probs overwrite this tile's q rows
#pragma unroll
    for (int r = 0; r < 8; ++r) {
      float mx = -1e30f;
#pragma unroll
      for (int nt = 0; nt < 4; ++nt) mx = fmaxf(mx, s[nt][r]);
      mx = redmax16(mx);
      float e4[4], sum = 0.f;
#pragma unroll
      for (int nt = 0; nt < 4; ++nt) {
        e4[nt] = __expf(0.125f * (s[nt][r] - mx));   // scale = 1/sqrt(64)
        sum += e4[nt];
      }
      sum = redsum(sum, 8);
      float inv = 1.f / sum;
      int seq = mt * 16 + r + 8 * hf;
#pragma unroll
      for (int nt = 0; nt < 4; ++nt)
        P[seq * 64 + nt * 16 + (lane & 15)] = f2bf(e4[nt] * inv);
    }
    // o = P @ V  (B-fragment from row-major v^T[d][seq])
#pragma unroll
    for (int nt = 0; nt < 4; ++nt) {
      floatx8 c = zero8();
#pragma unroll
      for (int ks = 0; ks < 2; ++ks)
        c = wmma_bf16(lds_frag(P, 64, mt * 16, ks * 32),
                      lds_frag(vtL, 64, nt * 16, ks * 32), c);
#pragma unroll
      for (int r = 0; r < 8; ++r) {
        int seq = mt * 16 + r + 8 * hf;
        obuf[((size_t)b * 64 + seq) * 512 + head * 64 + nt * 16 + (lane & 15)] =
            f2bf(c[r]);
      }
    }
  }
}

// ---------------------------------------------------------------------------
// Kernel 3b: out-proj + residual + LN.  grid = B*4 = 64 (16-row tiles).
// ---------------------------------------------------------------------------
__global__ __launch_bounds__(256) void outproj_kernel(
    const unsigned short* __restrict__ obuf, const unsigned int* __restrict__ outF,
    const float* __restrict__ out_b, const float* __restrict__ ln_g,
    const float* __restrict__ ln_b, float* __restrict__ xbuf, int layer) {
  __shared__ unsigned short xo[16 * 512];
  __shared__ float y[16 * 512];
  int b = blockIdx.x >> 2, mt = blockIdx.x & 3;
  int tid = threadIdx.x, lane = tid & 31, w = tid >> 5, hf = lane >> 4;
  const unsigned int* of = outF + (size_t)layer * (512 * 512 / 2);
  const float* ob = out_b + layer * 512;
  const float* lg = ln_g + layer * 512;
  const float* lb = ln_b + layer * 512;
  const unsigned short* src = obuf + ((size_t)b * 64 + mt * 16) * 512;
  for (int e = tid; e < 16 * 512; e += 256) xo[e] = src[e];
  __syncthreads();
  for (int i = 0; i < 4; ++i) {
    int nt = w + 8 * i;
    floatx8 c = zero8();
#pragma unroll
    for (int ks = 0; ks < 16; ++ks)
      c = wmma_bf16(lds_frag(xo, 512, 0, ks * 32), gmem_frag(of, nt, ks, 16), c);
    int n = nt * 16 + (lane & 15);
    float bias = ob[n];
#pragma unroll
    for (int r = 0; r < 8; ++r) y[(r + 8 * hf) * 512 + n] = c[r] + bias;
  }
  __syncthreads();
  float* xr = xbuf + ((size_t)b * 64 + mt * 16) * 512;
  for (int rr = 0; rr < 2; ++rr) {
    int row = 2 * w + rr;
    float vals[16], s = 0.f, ss = 0.f;
#pragma unroll
    for (int j = 0; j < 16; ++j) {
      int f = lane + 32 * j;
      float v = y[row * 512 + f] + xr[row * 512 + f];   // residual
      vals[j] = v; s += v; ss += v * v;
    }
    s = redsum(s, 16); ss = redsum(ss, 16);
    float mean = s * (1.f / 512.f);
    float inv = rsqrtf(ss * (1.f / 512.f) - mean * mean + 1e-5f);
#pragma unroll
    for (int j = 0; j < 16; ++j) {
      int f = lane + 32 * j;
      xr[row * 512 + f] = (vals[j] - mean) * inv * lg[f] + lb[f];
    }
  }
}

// ---------------------------------------------------------------------------
// Kernel 4: utterance pooling + MLP head (tiny, plain fp32).  grid = B = 16.
// ---------------------------------------------------------------------------
__global__ __launch_bounds__(256) void head_kernel(
    const float* __restrict__ xbuf, const float* __restrict__ up_w,
    const float* __restrict__ up_b, const float* __restrict__ c1_w,
    const float* __restrict__ c1_b, const float* __restrict__ c2_w,
    const float* __restrict__ c2_b, float* __restrict__ out) {
  __shared__ float u[512];
  __shared__ float h1[256];
  __shared__ float h2[128];
  int b = blockIdx.x, tid = threadIdx.x;
  const float* xb = xbuf + (size_t)b * 64 * 512;
  for (int f = tid; f < 512; f += 256) {
    float s = 0.f;
    for (int r = 0; r < 64; ++r) s += xb[r * 512 + f];
    u[f] = s * (1.f / 64.f);
  }
  __syncthreads();
  {
    float s = up_b[tid];
    for (int k = 0; k < 512; ++k) s += u[k] * up_w[k * 256 + tid];
    h1[tid] = s;
  }
  __syncthreads();
  if (tid < 128) {
    float s = c1_b[tid];
    for (int k = 0; k < 256; ++k) s += h1[k] * c1_w[k * 128 + tid];
    h2[tid] = fmaxf(s, 0.f);
  }
  __syncthreads();
  if (tid < 2) {
    float s = c2_b[tid];
    for (int k = 0; k < 128; ++k) s += h2[k] * c2_w[k * 2 + tid];
    out[b * 2 + tid] = s;
  }
}

// ---------------------------------------------------------------------------
// Workspace layout (bytes)
// ---------------------------------------------------------------------------
#define OFF_FEF   0u                 // 128*512*2      = 131072
#define OFF_SPF   131072u            // 512*512*2      = 524288
#define OFF_QKVF  655360u            // 6*512*1536*2   = 9437184
#define OFF_OUTF  10092544u          // 6*512*512*2    = 3145728
#define OFF_SEG   13238272u          // 16*64*512*4    = 2097152
#define OFF_X     15335424u          // 16*64*512*4    = 2097152
#define OFF_OBUF  17432576u          // 16*64*512*2    = 1048576  -> end 18481152

extern "C" void kernel_launch(void* const* d_in, const int* in_sizes, int n_in,
                              void* d_out, int out_size, void* d_ws, size_t ws_size,
                              hipStream_t stream) {
  (void)in_sizes; (void)n_in; (void)out_size; (void)ws_size;
  const float* mel    = (const float*)d_in[0];
  // d_in[1] = chunk_lengths: unused by the reference computation
  const float* fe_w   = (const float*)d_in[2];
  const float* fe_b   = (const float*)d_in[3];
  const float* fe_g   = (const float*)d_in[4];
  const float* fe_be  = (const float*)d_in[5];
  const float* sp_w   = (const float*)d_in[6];
  const float* sp_b   = (const float*)d_in[7];
  const float* sp_g   = (const float*)d_in[8];
  const float* sp_be  = (const float*)d_in[9];
  const float* qkv_w  = (const float*)d_in[10];
  const float* qkv_b  = (const float*)d_in[11];
  const float* out_w  = (const float*)d_in[12];
  const float* out_b  = (const float*)d_in[13];
  const float* ln_g   = (const float*)d_in[14];
  const float* ln_b   = (const float*)d_in[15];
  const float* up_w   = (const float*)d_in[16];
  const float* up_b   = (const float*)d_in[17];
  const float* c1_w   = (const float*)d_in[18];
  const float* c1_b   = (const float*)d_in[19];
  const float* c2_w   = (const float*)d_in[20];
  const float* c2_b   = (const float*)d_in[21];

  char* ws = (char*)d_ws;
  unsigned int*   feF  = (unsigned int*)(ws + OFF_FEF);
  unsigned int*   spF  = (unsigned int*)(ws + OFF_SPF);
  unsigned int*   qkvF = (unsigned int*)(ws + OFF_QKVF);
  unsigned int*   outF = (unsigned int*)(ws + OFF_OUTF);
  float*          segb = (float*)(ws + OFF_SEG);
  float*          xbuf = (float*)(ws + OFF_X);
  unsigned short* obuf = (unsigned short*)(ws + OFF_OBUF);

  // Pre-swizzle all weights into WMMA B-fragment order (bf16).
  prep_frag<<<(128 * 512 / 2 + 255) / 256, 256, 0, stream>>>(fe_w, feF, 128, 512);
  prep_frag<<<(512 * 512 / 2 + 255) / 256, 256, 0, stream>>>(sp_w, spF, 512, 512);
  for (int l = 0; l < 6; ++l) {
    prep_frag<<<(512 * 1536 / 2) / 256, 256, 0, stream>>>(
        qkv_w + (size_t)l * 512 * 1536, qkvF + (size_t)l * 512 * 1536 / 2, 512, 1536);
    prep_frag<<<(512 * 512 / 2) / 256, 256, 0, stream>>>(
        out_w + (size_t)l * 512 * 512, outF + (size_t)l * 512 * 512 / 2, 512, 512);
  }

  // seg padded rows 60..63 must be zero.
  zero_kernel<<<(16 * 64 * 512) / 256, 256, 0, stream>>>(segb);

  encoder_kernel<<<960, 256, 0, stream>>>(mel, feF, fe_b, fe_g, fe_be, segb);
  segproj_kernel<<<64, 256, 0, stream>>>(segb, spF, sp_b, sp_g, sp_be, xbuf);
  for (int l = 0; l < 6; ++l) {
    attn_kernel<<<128, 256, 90112, stream>>>(xbuf, qkvF, qkv_b, obuf, l);
    outproj_kernel<<<64, 256, 0, stream>>>(obuf, outF, out_b, ln_g, ln_b, xbuf, l);
  }
  head_kernel<<<16, 256, 0, stream>>>(xbuf, up_w, up_b, c1_w, c1_b, c2_w, c2_b,
                                      (float*)d_out);
}
